// SynthesisBlock_84670985274098
// MI455X (gfx1250) — compile-verified, matrix operand order checked by reading, not run on hardware
//
#include <hip/hip_runtime.h>
#include <hip/hip_bf16.h>
#include <stdint.h>

// ---------------------------------------------------------------------------
// CDNA5 / gfx1250 WMMA types
// ---------------------------------------------------------------------------
typedef __attribute__((ext_vector_type(16))) __bf16 v16bf;
typedef __attribute__((ext_vector_type(8)))  float  v8f;

union FragBF {
    uint4 q[2];   // 2 x 16B = 8 dwords = 16 bf16
    v16bf v;
};

__device__ __forceinline__ unsigned short f2bf(float f) {
    unsigned u = __builtin_bit_cast(unsigned, f);
    u += 0x7FFFu + ((u >> 16) & 1u);     // round-to-nearest-even
    return (unsigned short)(u >> 16);
}

// ---------------------------------------------------------------------------
// Problem constants (from reference setup_inputs)
// ---------------------------------------------------------------------------
#define BATCH 8
#define IMG   256          // output H=W
#define CIN0  128
#define COUT  64
#define WDIM  512

// ---------------------------------------------------------------------------
// Kernel 0: zero a small pad page used for SAME-padding OOB loads
// ---------------------------------------------------------------------------
__global__ void zero_page(uint4* p) {
    p[threadIdx.x] = uint4{0u, 0u, 0u, 0u};   // 256 * 16B = 4096B
}

// ---------------------------------------------------------------------------
// Kernel 1: styles + demodulation scale  g[b,co] = s * rsqrt(s^2*sumsq + eps)
// 1024 threads total: set(2) x b(8) x co(64)
// ---------------------------------------------------------------------------
__global__ void styles_kernel(const float* __restrict__ wlat,
                              const float* __restrict__ y0W, const float* __restrict__ y0b,
                              const float* __restrict__ y1W, const float* __restrict__ y1b,
                              const float* __restrict__ conv0w, const float* __restrict__ conv1w,
                              float* __restrict__ g0, float* __restrict__ g1) {
    int tid = blockIdx.x * 256 + threadIdx.x;
    int set = tid >> 9;
    int b   = (tid >> 6) & 7;
    int co  = tid & 63;
    const float* yW = set ? y1W : y0W;
    const float* yb = set ? y1b : y0b;
    const float* cw = set ? conv1w : conv0w;
    int KK = set ? (9 * COUT) : (9 * CIN0);

    float s = yb[co];
    for (int k = 0; k < WDIM; ++k)
        s += wlat[b * WDIM + k] * yW[k * COUT + co];

    float q = 0.f;
    for (int k = 0; k < KK; ++k) {
        float v = cw[k * COUT + co];
        q += v * v;
    }
    float d = rsqrtf(s * s * q + 1e-8f);
    (set ? g1 : g0)[b * COUT + co] = s * d;
}

// ---------------------------------------------------------------------------
// Kernel 2: pack conv weights (f32 [3,3,Cin,64]) into bf16 WMMA B-fragments.
// dword index = ((ks*4 + nt)*32 + lane)*8 + d, ks = kpos*(Cin/32)+cs.
// Lane layout mirrors the ISA 16-bit fragment table:
//   lanes 0-15  hold K-slice offsets {0..7, 16..23}
//   lanes 16-31 hold K-slice offsets {8..15, 24..31}
// ---------------------------------------------------------------------------
__global__ void pack_weights(const float* __restrict__ wsrc, int Cin,
                             unsigned int* __restrict__ dst, int total) {
    int idx = blockIdx.x * 256 + threadIdx.x;
    if (idx >= total) return;
    int d    = idx & 7;
    int lane = (idx >> 3) & 31;
    int nt   = (idx >> 8) & 3;
    int ks   = idx >> 10;
    int csteps = Cin >> 5;
    int kpos = ks / csteps;
    int cs   = ks % csteps;
    int hh   = lane >> 4;
    int co   = nt * 16 + (lane & 15);
    int base = ((d < 4) ? 0 : 16) + hh * 8;
    int k0   = cs * 32 + base + (d & 3) * 2;     // channel index within Cin
    float w0 = wsrc[(long)(kpos * Cin + k0)     * COUT + co];
    float w1 = wsrc[(long)(kpos * Cin + k0 + 1) * COUT + co];
    dst[idx] = (unsigned)f2bf(w0) | ((unsigned)f2bf(w1) << 16);
}

// ---------------------------------------------------------------------------
// Kernel 3: half-pixel bilinear 2x upsample, f32 -> bf16, NHWC.
// One thread = 4 channels of one output pixel. 16.7M threads.
// ---------------------------------------------------------------------------
__global__ void upsample_bf16(const float* __restrict__ x, unsigned short* __restrict__ out) {
    int idx = blockIdx.x * 256 + threadIdx.x;
    int c4 = idx & 31;
    int j  = (idx >> 5)  & 255;
    int i  = (idx >> 13) & 255;
    int b  = idx >> 21;

    int i0; float fi;
    if (i & 1) { i0 = (i - 1) >> 1; fi = 0.25f; } else { i0 = (i >> 1) - 1; fi = 0.75f; }
    int j0; float fj;
    if (j & 1) { j0 = (j - 1) >> 1; fj = 0.25f; } else { j0 = (j >> 1) - 1; fj = 0.75f; }
    int i1 = i0 + 1, j1 = j0 + 1;
    i0 = i0 < 0 ? 0 : i0;  j0 = j0 < 0 ? 0 : j0;
    i1 = i1 > 127 ? 127 : i1;  j1 = j1 > 127 ? 127 : j1;

    const float4* xv = (const float4*)x;   // [8,128,128,32] float4
    float4 a00 = xv[(((b * 128 + i0) * 128) + j0) * 32 + c4];
    float4 a01 = xv[(((b * 128 + i0) * 128) + j1) * 32 + c4];
    float4 a10 = xv[(((b * 128 + i1) * 128) + j0) * 32 + c4];
    float4 a11 = xv[(((b * 128 + i1) * 128) + j1) * 32 + c4];

    float w00 = (1.f - fi) * (1.f - fj), w01 = (1.f - fi) * fj;
    float w10 = fi * (1.f - fj),         w11 = fi * fj;

    float r0 = w00 * a00.x + w01 * a01.x + w10 * a10.x + w11 * a11.x;
    float r1 = w00 * a00.y + w01 * a01.y + w10 * a10.y + w11 * a11.y;
    float r2 = w00 * a00.z + w01 * a01.z + w10 * a10.z + w11 * a11.z;
    float r3 = w00 * a00.w + w01 * a01.w + w10 * a10.w + w11 * a11.w;

    uint2 pk;
    pk.x = (unsigned)f2bf(r0) | ((unsigned)f2bf(r1) << 16);
    pk.y = (unsigned)f2bf(r2) | ((unsigned)f2bf(r3) << 16);
    ((uint2*)out)[(((b * 256 + i) * 256) + j) * 32 + c4] = pk;
}

// ---------------------------------------------------------------------------
// Kernel 4: implicit-GEMM 3x3 SAME conv via v_wmma_f32_16x16x32_bf16 + fused
// epilogue (demod scale, noise*scale, bias, leaky ReLU).
// Block = 256 threads = 8 waves; wave = 64-pixel x 64-cout tile (M-block=4):
// each B fragment is loaded from LDS once and feeds 4 WMMAs (wmma:ds = 2:1).
// All packed weights staged in LDS (conv0: 147456B < 320KB WGP LDS), amortized
// over 512 output pixels per block (1024 blocks total).
// ---------------------------------------------------------------------------
template <int CIN, bool OUTBF>
__global__ void __launch_bounds__(256)
conv_wmma(const unsigned short* __restrict__ in,     // bf16 [B,256,256,CIN]
          const unsigned int*   __restrict__ wpack,  // packed bf16 B-fragments
          const float* __restrict__ g,               // [B,64] demod scales
          const float* __restrict__ noise,           // [B,256,256,1]
          const float* __restrict__ nscale,          // [B,256,256,64]
          const float* __restrict__ bias,            // [64]
          unsigned short* __restrict__ out_bf,       // bf16 out (conv0)
          float* __restrict__ out_f32,               // f32 out (conv1)
          const unsigned short* __restrict__ zeropad) {
    constexpr int CSTEPS = CIN / 32;
    constexpr int LDSDW  = 9 * CSTEPS * 4 * 32 * 8;   // dwords of packed weights
    extern __shared__ unsigned int lds[];

    const int tid = threadIdx.x;
    // Cooperative weight stage: global -> LDS (b128 traffic)
    {
        const uint4* src4 = (const uint4*)wpack;
        uint4* dst4 = (uint4*)lds;
        for (int t = tid; t < LDSDW / 4; t += 256) dst4[t] = src4[t];
    }
    __syncthreads();

    const int lane = tid & 31;
    const int wave = tid >> 5;
    const int hh   = lane >> 4;       // K-half select for A fragment
    const int mrow = lane & 15;       // pixel row M within a 16-pixel tile

    // Wave owns 64 consecutive pixels of one image row (4 M-tiles of 16).
    const long pbase = (long)blockIdx.x * 512 + wave * 64;    // flat pixel index
    const int b  = (int)(pbase >> 16);                        // /(256*256)
    const int hw = (int)(pbase & 65535);
    const int h  = hw >> 8;
    const int w0 = hw & 255;            // 0/64/128/192: strip stays in-row

    v8f acc[4][4] = {};                 // [mtile][ntile], 128 VGPRs

#pragma unroll 1
    for (int kpos = 0; kpos < 9; ++kpos) {
        const int dh = kpos / 3 - 1, dw = kpos % 3 - 1;
        const int hs = h + dh;
        const unsigned short* rowp[4];
        bool oob[4];
#pragma unroll
        for (int mt = 0; mt < 4; ++mt) {
            const int wsx = w0 + mt * 16 + mrow + dw;
            oob[mt]  = ((unsigned)hs >= 256u) | ((unsigned)wsx >= 256u);
            rowp[mt] = in + (((long)b * 256 + hs) * 256 + wsx) * CIN;
        }
        if (!oob[0]) __builtin_prefetch(rowp[0] + CIN, 0, 1);   // global_prefetch_b8
#pragma unroll
        for (int cs = 0; cs < CSTEPS; ++cs) {
            const int c0 = cs * 32 + hh * 8;
            const int ks = kpos * CSTEPS + cs;
            // B fragments for all 4 cout tiles: loaded once, used 4x each.
            FragBF bm[4];
#pragma unroll
            for (int nt = 0; nt < 4; ++nt) {
                const uint4* bp = (const uint4*)&lds[(((ks * 4 + nt) * 32 + lane) * 8)];
                bm[nt].q[0] = bp[0];
                bm[nt].q[1] = bp[1];
            }
#pragma unroll
            for (int mt = 0; mt < 4; ++mt) {
                FragBF a;
                const uint4* s0 = oob[mt] ? (const uint4*)zeropad : (const uint4*)(rowp[mt] + c0);
                const uint4* s1 = oob[mt] ? (const uint4*)zeropad : (const uint4*)(rowp[mt] + c0 + 16);
                a.q[0] = s0[0];
                a.q[1] = s1[0];
#pragma unroll
                for (int nt = 0; nt < 4; ++nt) {
                    acc[mt][nt] = __builtin_amdgcn_wmma_f32_16x16x32_bf16(
                        false, a.v, false, bm[nt].v, (short)0, acc[mt][nt], false, false);
                }
            }
        }
    }

    // Fused epilogue: demod scale + noise*scale + bias + leaky ReLU
    float lg[4], bs[4];
#pragma unroll
    for (int nt = 0; nt < 4; ++nt) {
        const int co = nt * 16 + (lane & 15);
        lg[nt] = g[b * COUT + co];
        bs[nt] = bias[co];
    }
#pragma unroll
    for (int mt = 0; mt < 4; ++mt) {
        const long pm = pbase + mt * 16;
#pragma unroll
        for (int r = 0; r < 8; ++r) {
            const long p = pm + r + hh * 8;      // C/D layout: lanes16-31 -> M=r+8
            const float nz = noise[p];
#pragma unroll
            for (int nt = 0; nt < 4; ++nt) {
                const int co = nt * 16 + (lane & 15);
                float v = acc[mt][nt][r] * lg[nt] + nscale[p * COUT + co] * nz + bs[nt];
                v = v > 0.f ? v : 0.2f * v;
                if (OUTBF) out_bf[p * COUT + co] = f2bf(v);
                else       out_f32[p * COUT + co] = v;
            }
        }
    }
}

// ---------------------------------------------------------------------------
// Launcher
// ---------------------------------------------------------------------------
extern "C" void kernel_launch(void* const* d_in, const int* in_sizes, int n_in,
                              void* d_out, int out_size, void* d_ws, size_t ws_size,
                              hipStream_t stream) {
    (void)in_sizes; (void)n_in; (void)out_size; (void)ws_size;
    const float* x     = (const float*)d_in[0];
    const float* wlat  = (const float*)d_in[1];
    const float* noise = (const float*)d_in[2];
    const float* c0w   = (const float*)d_in[3];
    const float* c1w   = (const float*)d_in[4];
    const float* y0W   = (const float*)d_in[5];
    const float* y0b   = (const float*)d_in[6];
    const float* y1W   = (const float*)d_in[7];
    const float* y1b   = (const float*)d_in[8];
    const float* ns0   = (const float*)d_in[9];
    const float* ns1   = (const float*)d_in[10];
    const float* bias0 = (const float*)d_in[11];

    // Workspace layout
    char* ws = (char*)d_ws;
    unsigned short* zeropad = (unsigned short*)(ws);                 //   4 KB (zeroed)
    float*          g0      = (float*)(ws + 4096);                   //   2 KB
    float*          g1      = (float*)(ws + 8192);                   //   2 KB
    unsigned int*   wp0     = (unsigned int*)(ws + 16384);           // 147456 B
    unsigned int*   wp1     = (unsigned int*)(ws + 16384 + 147456);  //  73728 B
    unsigned short* xup     = (unsigned short*)(ws + 262144);        // 134217728 B bf16
    unsigned short* y0mid   = (unsigned short*)(ws + 262144 + 134217728); // 67108864 B bf16

    zero_page<<<1, 256, 0, stream>>>((uint4*)zeropad);
    styles_kernel<<<4, 256, 0, stream>>>(wlat, y0W, y0b, y1W, y1b, c0w, c1w, g0, g1);

    const int tot0 = 9 * (CIN0 / 32) * 4 * 32 * 8;   // 36864 dwords
    const int tot1 = 9 * (COUT / 32) * 4 * 32 * 8;   // 18432 dwords
    pack_weights<<<(tot0 + 255) / 256, 256, 0, stream>>>(c0w, CIN0, wp0, tot0);
    pack_weights<<<(tot1 + 255) / 256, 256, 0, stream>>>(c1w, COUT, wp1, tot1);

    upsample_bf16<<<65536, 256, 0, stream>>>(x, xup);

    const int nblocks = BATCH * IMG * IMG / 512;     // 1024 blocks, 8 waves each
    conv_wmma<CIN0, true><<<nblocks, 256, tot0 * 4, stream>>>(
        xup, wp0, g0, noise, ns0, bias0, y0mid, nullptr, zeropad);
    conv_wmma<COUT, false><<<nblocks, 256, tot1 * 4, stream>>>(
        y0mid, wp1, g1, noise, ns1, bias0, nullptr, (float*)d_out, zeropad);
}
